// SelfAttn_5970004541837
// MI455X (gfx1250) — matrix operand order repaired and checked by reading, not compiled, not run on hardware
//
#include <hip/hip_runtime.h>
#include <hip/hip_bf16.h>
#include <stdint.h>

// ---------------------------------------------------------------------------
// Self-attention with relative position bias, fused for MI455X (gfx1250).
// B=4, L=1024, D=1024, H=16, ADIM=64, K_REL=8.
// Compute-bound (~52 GFLOP vs ~60MB HBM): every matmul routed through
// v_wmma_f32_16x16x32_f16; f16-resident tiles staged with
// global_load_async_to_lds_b128; all loops double-buffered; B-fragments
// batched into distinct registers so ds_load_b128 issues ahead of the
// WMMA chain (staggered s_wait_dscnt instead of hard 0 waits).
// ---------------------------------------------------------------------------

#define B_    4
#define L_    1024
#define D_    1024
#define H_    16
#define ADIM  64
#define KREL  8

typedef __attribute__((ext_vector_type(16))) _Float16 v16h;
typedef __attribute__((ext_vector_type(8)))  float    v8f;

union Frag16 { v16h v; uint32_t u[8]; _Float16 h[16]; };
union Pack8  { uint4 u; _Float16 h[8]; };
union Pack4  { uint2 u; _Float16 h[4]; };

static __device__ __forceinline__ v8f wmma_f16(v16h a, v16h b, v8f c) {
  // D = A(16x32 f16) x B(32x16 f16) + C(16x16 f32)
  return __builtin_amdgcn_wmma_f32_16x16x32_f16(
      /*neg_a=*/false, a, /*neg_b=*/false, b,
      /*c_mod=*/(short)0, c, /*reuse_a=*/false, /*reuse_b=*/false);
}

// A-matrix fragment: row-major LDS row of >=32 f16, 4-byte aligned.
// Contiguous K order (matches load_B): VGPR j holds K pair kb,kb+1 with
// kb = 2*j + 16*(lane>=16).
static __device__ __forceinline__ v16h load_A(const _Float16* row, int hi) {
  Frag16 f;
  const uint32_t* r32 = (const uint32_t*)row;
#pragma unroll
  for (int j = 0; j < 8; ++j) {
    int kb = 2 * j + 16 * hi;  // always even
    f.u[j] = r32[kb >> 1];
  }
  return f.v;
}

// B-matrix fragment: 16 contiguous f16 (K-run) per lane, 4-byte aligned.
static __device__ __forceinline__ v16h load_B(const _Float16* p) {
  Frag16 f;
  const uint32_t* r32 = (const uint32_t*)p;
#pragma unroll
  for (int j = 0; j < 8; ++j) f.u[j] = r32[j];
  return f.v;
}

// Async global->LDS copy, 16B per lane, tracked on ASYNCcnt.
// LDS byte offset = low 32 bits of the flat shared-pointer (aperture rule).
static __device__ __forceinline__ void async_b128(void* lds, const void* g) {
  asm volatile("global_load_async_to_lds_b128 %0, %1, off"
               :: "v"((uint32_t)(uintptr_t)lds), "v"(g)
               : "memory");
}
static __device__ __forceinline__ void wait_async0() {
  asm volatile("s_wait_asynccnt 0x0" ::: "memory");
}
static __device__ __forceinline__ void wait_async4() {
  asm volatile("s_wait_asynccnt 0x4" ::: "memory");  // retire all but newest 4
}

// ---------------------------------------------------------------------------
// Kernel 1: fused QKV projection.
// Y[4096,3072] = X[4096,1024] @ W^T + b, scattered into Qf/Kf/Vf as
// [B,H,L,ADIM] f16. Block 128 thr (4 waves); tile M128 x N64; K-step 32;
// double-buffered LDS; 8 WMMA/wave/k-step (2 A-frags share 4 B-frags).
// ---------------------------------------------------------------------------
__global__ __launch_bounds__(128) void qkv_gemm_kernel(
    const float* __restrict__ X, const float* __restrict__ W,
    const float* __restrict__ bias,
    _Float16* __restrict__ Qf, _Float16* __restrict__ Kf,
    _Float16* __restrict__ Vf) {
  __shared__ _Float16 At[2][128][32];
  __shared__ _Float16 Bt[2][64][32];

  const int tid  = threadIdx.x;
  const int w    = tid >> 5;
  const int lane = tid & 31;
  const int ln16 = lane & 15;
  const int hi   = lane >> 4;
  const int m0   = blockIdx.y * 128;  // [0,4096)
  const int n0   = blockIdx.x * 64;   // [0,3072)

  v8f acc[2][4] = {};

  const float4* Xg = (const float4*)X;  // row stride 256 float4
  const float4* Wg = (const float4*)W;

  // Prologue: prefetch tile 0 into registers.
  float4 ax[8], wx[4];
#pragma unroll
  for (int i = 0; i < 8; ++i) {
    int e = tid + i * 128, row = e >> 3, q = e & 7;
    ax[i] = Xg[(size_t)(m0 + row) * 256 + q];
  }
#pragma unroll
  for (int i = 0; i < 4; ++i) {
    int e = tid + i * 128, row = e >> 3, q = e & 7;
    wx[i] = Wg[(size_t)(n0 + row) * 256 + q];
  }

  int cur = 0;
  for (int kt = 0; kt < D_; kt += 32) {
    // Convert & commit current tile to LDS[cur].
#pragma unroll
    for (int i = 0; i < 8; ++i) {
      int e = tid + i * 128, row = e >> 3, q = e & 7;
      Pack4 p;
      p.h[0] = (_Float16)ax[i].x; p.h[1] = (_Float16)ax[i].y;
      p.h[2] = (_Float16)ax[i].z; p.h[3] = (_Float16)ax[i].w;
      *(uint2*)&At[cur][row][q * 4] = p.u;
    }
#pragma unroll
    for (int i = 0; i < 4; ++i) {
      int e = tid + i * 128, row = e >> 3, q = e & 7;
      Pack4 p;
      p.h[0] = (_Float16)wx[i].x; p.h[1] = (_Float16)wx[i].y;
      p.h[2] = (_Float16)wx[i].z; p.h[3] = (_Float16)wx[i].w;
      *(uint2*)&Bt[cur][row][q * 4] = p.u;
    }
    // Prefetch next k-slice (loads overlap this step's WMMA).
    if (kt + 32 < D_) {
      int ko = (kt >> 2) + 8;
#pragma unroll
      for (int i = 0; i < 8; ++i) {
        int e = tid + i * 128, row = e >> 3, q = e & 7;
        ax[i] = Xg[(size_t)(m0 + row) * 256 + ko + q];
      }
#pragma unroll
      for (int i = 0; i < 4; ++i) {
        int e = tid + i * 128, row = e >> 3, q = e & 7;
        wx[i] = Wg[(size_t)(n0 + row) * 256 + ko + q];
      }
    }
    __syncthreads();  // single barrier/iter: next stores hit the other buffer

    // All fragments live at once -> ds_load batch, then WMMA chain.
    v16h a0 = load_A(&At[cur][w * 32 + ln16][0], hi);
    v16h a1 = load_A(&At[cur][w * 32 + 16 + ln16][0], hi);
    v16h b0 = load_B(&Bt[cur][0  + ln16][hi * 16]);
    v16h b1 = load_B(&Bt[cur][16 + ln16][hi * 16]);
    v16h b2 = load_B(&Bt[cur][32 + ln16][hi * 16]);
    v16h b3 = load_B(&Bt[cur][48 + ln16][hi * 16]);
    acc[0][0] = wmma_f16(a0, b0, acc[0][0]);
    acc[1][0] = wmma_f16(a1, b0, acc[1][0]);
    acc[0][1] = wmma_f16(a0, b1, acc[0][1]);
    acc[1][1] = wmma_f16(a1, b1, acc[1][1]);
    acc[0][2] = wmma_f16(a0, b2, acc[0][2]);
    acc[1][2] = wmma_f16(a1, b2, acc[1][2]);
    acc[0][3] = wmma_f16(a0, b3, acc[0][3]);
    acc[1][3] = wmma_f16(a1, b3, acc[1][3]);
    cur ^= 1;
  }

  // Epilogue: bias add + scatter to Q/K/V [B,H,L,ADIM] f16.
#pragma unroll
  for (int sub = 0; sub < 2; ++sub) {
#pragma unroll
    for (int nb = 0; nb < 4; ++nb) {
      int col = n0 + nb * 16 + ln16;  // [0,3072)
      int s = col >> 10, rem = col & 1023;
      int hh = rem >> 6, d = rem & 63;
      _Float16* dst = (s == 0) ? Qf : (s == 1) ? Kf : Vf;
      float bv = bias[col];
#pragma unroll
      for (int r = 0; r < 8; ++r) {
        int row = m0 + w * 32 + sub * 16 + r + 8 * hi;  // [0,4096)
        int bi = row >> 10, l = row & 1023;
        dst[((size_t)(bi * H_ + hh) * L_ + l) * ADIM + d] =
            (_Float16)(acc[sub][nb][r] + bv);
      }
    }
  }
}

// ---------------------------------------------------------------------------
// Kernel 2: flash attention with relative-position bias + mask.
// Grid: (L/64, B*H); block 128 thr (4 waves, 16 q-rows each).
// Q/K tiles staged with async global->LDS; K/V double-buffered.
// ---------------------------------------------------------------------------
__global__ __launch_bounds__(128) void attn_kernel(
    const _Float16* __restrict__ Qf, const _Float16* __restrict__ Kf,
    const _Float16* __restrict__ Vf, const float* __restrict__ relE,
    const unsigned char* __restrict__ mask, _Float16* __restrict__ Of) {
  __shared__ _Float16 Qs[64 * 64];
  __shared__ _Float16 Ks[2][64 * 64];
  __shared__ _Float16 Vt[2][64][66];   // transposed, padded (bank-safe, 4B-aligned)
  __shared__ float    relb[64][17];    // per-(q-row, bucket) bias table
  __shared__ _Float16 Ps[4][16 * 64];  // per-wave P layout-crossover tile

  const int tid  = threadIdx.x;
  const int w    = tid >> 5;
  const int lane = tid & 31;
  const int ln16 = lane & 15;
  const int hi   = lane >> 4;
  const int bh   = blockIdx.y;  // b*H + h
  const int bi   = bh >> 4;
  const int h    = bh & 15;
  const int q0   = blockIdx.x * 64;

  const _Float16* Qh = Qf + (size_t)bh * L_ * ADIM;
  const _Float16* Kh = Kf + (size_t)bh * L_ * ADIM;
  const _Float16* Vh = Vf + (size_t)bh * L_ * ADIM;

  // Async-stage Q tile and K tile 0 (16B/lane x 4 each).
#pragma unroll
  for (int i = 0; i < 4; ++i) {
    int seg = tid + i * 128;
    async_b128((char*)Qs + seg * 16,
               (const char*)(Qh + (size_t)q0 * ADIM) + seg * 16);
  }
#pragma unroll
  for (int i = 0; i < 4; ++i) {
    int seg = tid + i * 128;
    async_b128((char*)Ks[0] + seg * 16, (const char*)Kh + seg * 16);
  }
  // V tile 0 into registers (transposed at LDS-store time).
  uint4 vx[4];
#pragma unroll
  for (int i = 0; i < 4; ++i) vx[i] = ((const uint4*)Vh)[tid + i * 128];

  wait_async4();   // Q copies (oldest 4) retired; K0 still in flight
  __syncthreads();

  // Rel-bias table: relb[row][r] = q_row . rel_emb[r]  (17 buckets).
  for (int e = tid; e < 64 * 17; e += 128) {
    int row = e / 17, ri = e % 17;
    float sacc = 0.f;
#pragma unroll
    for (int k2 = 0; k2 < ADIM; ++k2)
      sacc += (float)Qs[row * 64 + k2] * relE[ri * ADIM + k2];
    relb[row][ri] = sacc;
  }
  // Commit V tile 0 transposed.
#pragma unroll
  for (int i = 0; i < 4; ++i) {
    int seg = tid + i * 128, kp = seg >> 3, q = seg & 7;
    Pack8 p; p.u = vx[i];
#pragma unroll
    for (int j = 0; j < 8; ++j) Vt[0][q * 8 + j][kp] = p.h[j];
  }
  wait_async0();   // K0 resident
  __syncthreads();

  // Persistent Q A-fragments (K-dim = feature dim, two 16x32 slices).
  v16h aq0 = load_A(&Qs[(w * 16 + ln16) * 64 + 0], hi);
  v16h aq1 = load_A(&Qs[(w * 16 + ln16) * 64 + 32], hi);

  v8f acc_o[4] = {};
  float mst[8], lst[8];
#pragma unroll
  for (int r = 0; r < 8; ++r) { mst[r] = -1e30f; lst[r] = 0.f; }

  int cur = 0;
  for (int kt = 0; kt < L_; kt += 64) {
    const bool more = (kt + 64 < L_);
    if (more) {  // prefetch next K (async LDS) + next V (regs)
#pragma unroll
      for (int i = 0; i < 4; ++i) {
        int seg = tid + i * 128;
        async_b128((char*)Ks[cur ^ 1] + seg * 16,
                   (const char*)(Kh + (size_t)(kt + 64) * ADIM) + seg * 16);
      }
#pragma unroll
      for (int i = 0; i < 4; ++i)
        vx[i] = ((const uint4*)(Vh + (size_t)(kt + 64) * ADIM))[tid + i * 128];
    }

    // S = Q . K^T  (16x64 per wave, 8 WMMA), B-frags batched 4-at-a-time.
    v8f sacc[4] = {};
#pragma unroll
    for (int half = 0; half < 2; ++half) {
      const int nb0 = half * 2, nb1 = nb0 + 1;
      v16h b00 = load_B(&Ks[cur][(nb0 * 16 + ln16) * 64 + hi * 16]);
      v16h b01 = load_B(&Ks[cur][(nb0 * 16 + ln16) * 64 + 32 + hi * 16]);
      v16h b10 = load_B(&Ks[cur][(nb1 * 16 + ln16) * 64 + hi * 16]);
      v16h b11 = load_B(&Ks[cur][(nb1 * 16 + ln16) * 64 + 32 + hi * 16]);
      sacc[nb0] = wmma_f16(aq0, b00, sacc[nb0]);
      sacc[nb0] = wmma_f16(aq1, b01, sacc[nb0]);
      sacc[nb1] = wmma_f16(aq0, b10, sacc[nb1]);
      sacc[nb1] = wmma_f16(aq1, b11, sacc[nb1]);
    }

    // Bias + scale + mask; row-max.
    float mx[8];
#pragma unroll
    for (int r = 0; r < 8; ++r) mx[r] = -1e30f;
#pragma unroll
    for (int nb = 0; nb < 4; ++nb) {
      int colg = kt + nb * 16 + ln16;
#pragma unroll
      for (int r = 0; r < 8; ++r) {
        int rowl = w * 16 + r + 8 * hi;
        int rowg = q0 + rowl;
        int dlt = colg - rowg;
        dlt = dlt < -KREL ? -KREL : (dlt > KREL ? KREL : dlt);
        float sv = (sacc[nb][r] + relb[rowl][dlt + KREL]) * 0.125f;
        if (mask[(size_t)bi * L_ * L_ + (size_t)rowg * L_ + colg]) sv = -1e9f;
        sacc[nb][r] = sv;
        mx[r] = fmaxf(mx[r], sv);
      }
    }
#pragma unroll
    for (int r = 0; r < 8; ++r) {
#pragma unroll
      for (int off = 1; off < 16; off <<= 1)
        mx[r] = fmaxf(mx[r], __shfl_xor(mx[r], off, 32));
    }

    // Online softmax update.
    float alpha[8];
#pragma unroll
    for (int r = 0; r < 8; ++r) {
      float mnew = fmaxf(mst[r], mx[r]);
      alpha[r] = __expf(mst[r] - mnew);
      mst[r] = mnew;
    }
    float rs[8];
#pragma unroll
    for (int r = 0; r < 8; ++r) rs[r] = 0.f;
#pragma unroll
    for (int nb = 0; nb < 4; ++nb)
#pragma unroll
      for (int r = 0; r < 8; ++r) {
        float p = __expf(sacc[nb][r] - mst[r]);
        sacc[nb][r] = p;
        rs[r] += p;
      }
#pragma unroll
    for (int r = 0; r < 8; ++r) {
#pragma unroll
      for (int off = 1; off < 16; off <<= 1)
        rs[r] += __shfl_xor(rs[r], off, 32);
      lst[r] = lst[r] * alpha[r] + rs[r];
#pragma unroll
      for (int db = 0; db < 4; ++db) acc_o[db][r] *= alpha[r];
    }

    // C-layout f32 -> A-layout f16 crossover via per-wave LDS tile.
    _Float16* ps = Ps[w];
#pragma unroll
    for (int nb = 0; nb < 4; ++nb) {
      int c = nb * 16 + ln16;
#pragma unroll
      for (int r = 0; r < 8; ++r)
        ps[(r + 8 * hi) * 64 + c] = (_Float16)sacc[nb][r];
    }
    asm volatile("s_wait_dscnt 0" ::: "memory");

    // O += P . V  (16x64 per wave, 8 WMMA), B-frags batched 4-at-a-time.
    v16h ap0 = load_A(&ps[ln16 * 64 + 0], hi);
    v16h ap1 = load_A(&ps[ln16 * 64 + 32], hi);
#pragma unroll
    for (int half = 0; half < 2; ++half) {
      const int db0 = half * 2, db1 = db0 + 1;
      const int d0 = db0 * 16 + ln16, d1 = db1 * 16 + ln16;
      v16h b00 = load_B(&Vt[cur][d0][hi * 16]);
      v16h b01 = load_B(&Vt[cur][d0][32 + hi * 16]);
      v16h b10 = load_B(&Vt[cur][d1][hi * 16]);
      v16h b11 = load_B(&Vt[cur][d1][32 + hi * 16]);
      acc_o[db0] = wmma_f16(ap0, b00, acc_o[db0]);
      acc_o[db0] = wmma_f16(ap1, b01, acc_o[db0]);
      acc_o[db1] = wmma_f16(ap0, b10, acc_o[db1]);
      acc_o[db1] = wmma_f16(ap1, b11, acc_o[db1]);
    }

    if (more) {  // commit next V tile (other buffer; overlapped loads retire here)
#pragma unroll
      for (int i = 0; i < 4; ++i) {
        int seg = tid + i * 128, kp = seg >> 3, q = seg & 7;
        Pack8 p; p.u = vx[i];
#pragma unroll
        for (int j = 0; j < 8; ++j) Vt[cur ^ 1][q * 8 + j][kp] = p.h[j];
      }
    }
    wait_async0();   // next K tile resident (copy ran under this iter's WMMA)
    __syncthreads();
    cur ^= 1;
  }

  // Normalize and store O as [B, L, H*ADIM] f16 (head-merged for out-proj).
#pragma unroll
  for (int r = 0; r < 8; ++r) {
    int rowg = q0 + w * 16 + r + 8 * hi;
    float inv = 1.f / fmaxf(lst[r], 1e-20f);
#pragma unroll
    for (int db = 0; db < 4; ++db) {
      int d = db * 16 + ln16;
      Of[((size_t)bi * L_ + rowg) * D_ + h * ADIM + d] =
          (_Float16)(acc_o[db][r] * inv);
    }
  }
}

// ---------------------------------------------------------------------------
// Kernel 3: output projection.  Out[4096,1024] = O_f16 @ W_out^T + b_out (f32).
// A tile (already f16) staged with async global->LDS, double-buffered.
// ---------------------------------------------------------------------------
__global__ __launch_bounds__(128) void out_gemm_kernel(
    const _Float16* __restrict__ Of, const float* __restrict__ W,
    const float* __restrict__ bias, float* __restrict__ Out) {
  __shared__ _Float16 At[2][128][32];
  __shared__ _Float16 Bt[2][64][32];

  const int tid  = threadIdx.x;
  const int w    = tid >> 5;
  const int lane = tid & 31;
  const int ln16 = lane & 15;
  const int hi   = lane >> 4;
  const int m0   = blockIdx.y * 128;
  const int n0   = blockIdx.x * 64;

  v8f acc[2][4] = {};
  const float4* Wg = (const float4*)W;

  // Prologue: async-stage A tile 0; W tile 0 into registers.
#pragma unroll
  for (int i = 0; i < 4; ++i) {
    int e = tid + i * 128, row = e >> 2, s4 = e & 3;
    async_b128(&At[0][row][s4 * 8], Of + (size_t)(m0 + row) * D_ + s4 * 8);
  }
  float4 wx[4];
#pragma unroll
  for (int i = 0; i < 4; ++i) {
    int e = tid + i * 128, row = e >> 3, q = e & 7;
    wx[i] = Wg[(size_t)(n0 + row) * 256 + q];
  }

  int cur = 0;
  for (int kt = 0; kt < D_; kt += 32) {
    // Commit W tile (f32->f16) to LDS[cur].
#pragma unroll
    for (int i = 0; i < 4; ++i) {
      int e = tid + i * 128, row = e >> 3, q = e & 7;
      Pack4 p;
      p.h[0] = (_Float16)wx[i].x; p.h[1] = (_Float16)wx[i].y;
      p.h[2] = (_Float16)wx[i].z; p.h[3] = (_Float16)wx[i].w;
      *(uint2*)&Bt[cur][row][q * 4] = p.u;
    }
    const bool more = (kt + 32 < D_);
    if (more) {  // prefetch next A (async) + next W (regs)
#pragma unroll
      for (int i = 0; i < 4; ++i) {
        int e = tid + i * 128, row = e >> 2, s4 = e & 3;
        async_b128(&At[cur ^ 1][row][s4 * 8],
                   Of + (size_t)(m0 + row) * D_ + (kt + 32) + s4 * 8);
      }
      int ko = (kt >> 2) + 8;
#pragma unroll
      for (int i = 0; i < 4; ++i) {
        int e = tid + i * 128, row = e >> 3, q = e & 7;
        wx[i] = Wg[(size_t)(n0 + row) * 256 + ko + q];
      }
      wait_async4();  // retire A[cur] (oldest 4, in-order); A[nxt] keeps flying
    } else {
      wait_async0();
    }
    __syncthreads();

    v16h a0 = load_A(&At[cur][w * 32 + ln16][0], hi);
    v16h a1 = load_A(&At[cur][w * 32 + 16 + ln16][0], hi);
    v16h b0 = load_B(&Bt[cur][0  + ln16][hi * 16]);
    v16h b1 = load_B(&Bt[cur][16 + ln16][hi * 16]);
    v16h b2 = load_B(&Bt[cur][32 + ln16][hi * 16]);
    v16h b3 = load_B(&Bt[cur][48 + ln16][hi * 16]);
    acc[0][0] = wmma_f16(a0, b0, acc[0][0]);
    acc[1][0] = wmma_f16(a1, b0, acc[1][0]);
    acc[0][1] = wmma_f16(a0, b1, acc[0][1]);
    acc[1][1] = wmma_f16(a1, b1, acc[1][1]);
    acc[0][2] = wmma_f16(a0, b2, acc[0][2]);
    acc[1][2] = wmma_f16(a1, b2, acc[1][2]);
    acc[0][3] = wmma_f16(a0, b3, acc[0][3]);
    acc[1][3] = wmma_f16(a1, b3, acc[1][3]);
    cur ^= 1;
  }

#pragma unroll
  for (int sub = 0; sub < 2; ++sub) {
#pragma unroll
    for (int nb = 0; nb < 4; ++nb) {
      int col = n0 + nb * 16 + ln16;
      float bv = bias[col];
#pragma unroll
      for (int r = 0; r < 8; ++r) {
        int row = m0 + w * 32 + sub * 16 + r + 8 * hi;
        Out[(size_t)row * D_ + col] = acc[sub][nb][r] + bv;
      }
    }
  }
}

// ---------------------------------------------------------------------------
// Host launcher.
// ---------------------------------------------------------------------------
extern "C" void kernel_launch(void* const* d_in, const int* in_sizes, int n_in,
                              void* d_out, int out_size, void* d_ws,
                              size_t ws_size, hipStream_t stream) {
  const float* iQ    = (const float*)d_in[0];
  const float* W_ad  = (const float*)d_in[1];
  const float* b_ad  = (const float*)d_in[2];
  const float* W_out = (const float*)d_in[3];
  const float* b_out = (const float*)d_in[4];
  const float* relE  = (const float*)d_in[5];
  const unsigned char* mask = (const unsigned char*)d_in[6];
  float* out = (float*)d_out;

  // Workspace layout (f16): Q,K,V as [B,H,L,ADIM]; O as [B,L,D].
  const size_t elems = (size_t)B_ * H_ * L_ * ADIM;  // 4,194,304
  char* ws = (char*)d_ws;
  _Float16* Qf = (_Float16*)(ws + 0 * elems * sizeof(_Float16));
  _Float16* Kf = (_Float16*)(ws + 1 * elems * sizeof(_Float16));
  _Float16* Vf = (_Float16*)(ws + 2 * elems * sizeof(_Float16));
  _Float16* Of = (_Float16*)(ws + 3 * elems * sizeof(_Float16));

  // 1) fused QKV projection: tiles N=3072/64, M=4096/128
  qkv_gemm_kernel<<<dim3(48, 32), 128, 0, stream>>>(iQ, W_ad, b_ad, Qf, Kf, Vf);
  // 2) flash attention with rel-pos bias + mask: (L/64, B*H)
  attn_kernel<<<dim3(16, 64), 128, 0, stream>>>(Qf, Kf, Vf, relE, mask, Of);
  // 3) output projection: tiles N=1024/64, M=4096/128
  out_gemm_kernel<<<dim3(16, 32), 128, 0, stream>>>(Of, W_out, b_out, out);
}
